// MoEFFN_73478300500024
// MI455X (gfx1250) — matrix operand order repaired and checked by reading, not compiled, not run on hardware
//
#include <hip/hip_runtime.h>

// ---------------- problem constants (from reference) ----------------
#define DIM    1024
#define HIDDEN 4096
#define NEXP   4
#define NTOK   8192          // B*S = 4*2048

// ---------------- types ----------------
typedef __attribute__((ext_vector_type(8)))  __bf16 bf16x8;
typedef __attribute__((ext_vector_type(16))) __bf16 bf16x16;
typedef __attribute__((ext_vector_type(8)))  float  v8f;

// ---------------- workspace layout (bytes) ----------------
// meta ints: [0..3]=counts  [4..7]=seg_off  [8..12]=tile_base(cum,5)  [13..16]=cursor
static constexpr size_t WS_META = 0;
static constexpr size_t WS_IDX  = 256;
static constexpr size_t WS_ORD  = WS_IDX + (size_t)NTOK * 4;
static constexpr size_t WS_XB   = 131072;                                   // x in bf16
static constexpr size_t WS_W1B  = WS_XB  + (size_t)NTOK * DIM * 2;
static constexpr size_t WS_W2B  = WS_W1B + (size_t)NEXP * DIM * HIDDEN * 2;
static constexpr size_t WS_H    = WS_W2B + (size_t)NEXP * HIDDEN * DIM * 2;
// total: WS_H + (NTOK+256)*HIDDEN*2  ~= 146 MB workspace required

// LDS tile geometry: double-buffered 64 x 128 bf16, pitch 136 (272B => 4-bank rot)
#define KCH   128
#define APITCH 136
#define ABUF  (64 * APITCH)

__device__ __forceinline__ float gelu_exact(float v) {
  return 0.5f * v * (1.0f + erff(v * 0.70710678118654752f));
}

// B frag: 16 contiguous bf16 per lane (pre-shuffled layout)
__device__ __forceinline__ bf16x16 ld_b_frag(const __bf16* p) {
  const bf16x8 lo = *(const bf16x8*)p;
  const bf16x8 hi = *(const bf16x8*)(p + 8);
  return __builtin_shufflevector(lo, hi, 0,1,2,3,4,5,6,7,8,9,10,11,12,13,14,15);
}
// A frag: 16 contiguous elems (row-major LDS tile)
__device__ __forceinline__ bf16x16 ld_a_frag(const __bf16* p) {
  const bf16x8 lo = *(const bf16x8*)p;
  const bf16x8 hi = *(const bf16x8*)(p + 8);
  return __builtin_shufflevector(lo, hi, 0,1,2,3,4,5,6,7,8,9,10,11,12,13,14,15);
}

// async 64B/thread copy global->LDS (4 x b128); offset applies to both sides
__device__ __forceinline__ void async_copy64B(uint32_t lds_off, uint64_t gaddr) {
  asm volatile("global_load_async_to_lds_b128 %0, %1, off offset:0\n\t"
               "global_load_async_to_lds_b128 %0, %1, off offset:16\n\t"
               "global_load_async_to_lds_b128 %0, %1, off offset:32\n\t"
               "global_load_async_to_lds_b128 %0, %1, off offset:48"
               :: "v"(lds_off), "v"(gaddr) : "memory");
}
__device__ __forceinline__ void async_wait4() {
  asm volatile("s_wait_asynccnt 0x4" ::: "memory");   // prev chunk retired
}
__device__ __forceinline__ void async_wait0() {
  asm volatile("s_wait_asynccnt 0x0" ::: "memory");
}

// ---------------- tiny setup kernels ----------------
__global__ void moe_init(int* meta) {
  if (threadIdx.x < 32) meta[threadIdx.x] = 0;
}

__global__ __launch_bounds__(256) void moe_router(
    const float* __restrict__ x, const float* __restrict__ rw,
    const float* __restrict__ rb, int* __restrict__ meta, int* __restrict__ idx) {
  const int lane = threadIdx.x & 31;
  const int t    = blockIdx.x * 8 + (threadIdx.x >> 5);
  const float* xr = x + (size_t)t * DIM;
  float a0 = 0.f, a1 = 0.f, a2 = 0.f, a3 = 0.f;
  for (int d = lane; d < DIM; d += 32) {
    const float  xv = xr[d];
    const float4 w  = *(const float4*)(rw + d * 4);   // router_w is [DIM, 4]
    a0 += xv * w.x; a1 += xv * w.y; a2 += xv * w.z; a3 += xv * w.w;
  }
#pragma unroll
  for (int m = 16; m >= 1; m >>= 1) {
    a0 += __shfl_xor(a0, m, 32);
    a1 += __shfl_xor(a1, m, 32);
    a2 += __shfl_xor(a2, m, 32);
    a3 += __shfl_xor(a3, m, 32);
  }
  if (lane == 0) {
    a0 += rb[0]; a1 += rb[1]; a2 += rb[2]; a3 += rb[3];
    int best = 0; float bv = a0;                 // softmax monotonic: argmax logits
    if (a1 > bv) { bv = a1; best = 1; }          // strict '>' keeps first max
    if (a2 > bv) { bv = a2; best = 2; }
    if (a3 > bv) { bv = a3; best = 3; }
    idx[t] = best;
    atomicAdd(&meta[best], 1);
  }
}

__global__ void moe_scan(int* meta) {
  if (threadIdx.x == 0 && blockIdx.x == 0) {
    int seg = 0, tb = 0;
    for (int e = 0; e < NEXP; ++e) {
      meta[4 + e]  = seg;
      meta[8 + e]  = tb;
      meta[13 + e] = 0;
      seg += meta[e];
      tb  += (meta[e] + 63) >> 6;   // 64-row tiles per expert
    }
    meta[12] = tb;                  // total tiles
  }
}

__global__ __launch_bounds__(256) void moe_scatter(
    const int* __restrict__ idx, int* __restrict__ meta, int* __restrict__ order) {
  const int t = blockIdx.x * 256 + threadIdx.x;
  if (t < NTOK) {
    const int e   = idx[t];
    const int pos = atomicAdd(&meta[13 + e], 1);
    order[meta[4 + e] + pos] = t;
  }
}

// ---------------- fp32 -> bf16 conversions / weight shuffles ----------------
__global__ __launch_bounds__(256) void moe_prep_x(
    const float* __restrict__ x, __bf16* __restrict__ xb) {
  const size_t i = (size_t)blockIdx.x * 256 + threadIdx.x;
  xb[i] = (__bf16)x[i];
}

// B frag (32x16 bf16, 16x16x32): lane<16 holds K=0..15 of col N=lane;
// lane>=16 holds K=16..31 of col N=lane-16; 16 contiguous bf16 per lane.
__global__ __launch_bounds__(256) void moe_prep_w1(
    const float* __restrict__ w1, __bf16* __restrict__ w1b) {
  const size_t i = (size_t)blockIdx.x * 256 + threadIdx.x;     // E*DIM*HIDDEN
  const int n = (int)(i & (HIDDEN - 1));
  const size_t ek = i >> 12;
  const int k = (int)(ek & (DIM - 1));
  const int e = (int)(i >> 22);
  const int kb = k >> 5, kr = k & 31, nb = n >> 4, nr = n & 15;
  const size_t o = ((((size_t)e * (DIM / 32) + kb) * (HIDDEN / 16) + nb) * 32
                    + (nr + (kr & 16))) * 16 + (kr & 15);
  w1b[o] = (__bf16)w1[i];
}

__global__ __launch_bounds__(256) void moe_prep_w2(
    const float* __restrict__ w2, __bf16* __restrict__ w2b) {
  const size_t i = (size_t)blockIdx.x * 256 + threadIdx.x;     // E*HIDDEN*DIM
  const int n = (int)(i & (DIM - 1));
  const size_t ek = i >> 10;
  const int k = (int)(ek & (HIDDEN - 1));
  const int e = (int)(i >> 22);
  const int kb = k >> 5, kr = k & 31, nb = n >> 4, nr = n & 15;
  const size_t o = ((((size_t)e * (HIDDEN / 32) + kb) * (DIM / 16) + nb) * 32
                    + (nr + (kr & 16))) * 16 + (kr & 15);
  w2b[o] = (__bf16)w2[i];
}

// ---------------- GEMM1: h = gelu(x_gathered @ w1[e] + b1[e]) ----------------
// block = 256 thr (8 waves). M-tile = 64 tokens, wave: 64x64 acc (4x4 frags).
// A double-buffered in LDS (K-chunks of 128) via async copy overlapped with MMA.
__global__ __launch_bounds__(256) void moe_gemm1(
    const __bf16* __restrict__ xb, const __bf16* __restrict__ w1b,
    const float* __restrict__ b1, const int* __restrict__ meta,
    const int* __restrict__ order, __bf16* __restrict__ hbuf) {
  __shared__ __bf16 As[2 * ABUF];
  const int tile = blockIdx.x;
  if (tile >= meta[12]) return;
  int e = 0;
#pragma unroll
  for (int q = 0; q < 3; ++q) if (tile >= meta[9 + q]) e = q + 1;
  const int ltile = tile - meta[8 + e];
  const int seg   = meta[4 + e];
  const int cnt   = meta[e];
  const int tid   = threadIdx.x;
  const int wave  = tid >> 5, lane = tid & 31;
  const int n0 = blockIdx.y * 512 + wave * 64;
  // staging: thread -> (row, 32-elem col group); gather token row once
  const int sr = tid >> 2;
  const int sc = (tid & 3) * 32;
  int rl = ltile * 64 + sr;
  if (rl >= cnt) rl = cnt - 1;                   // pad rows duplicate last token
  const int tok = order[seg + rl];
  const uint32_t lds0 = (uint32_t)(uintptr_t)(As + sr * APITCH + sc);
  const uint32_t lds1 = lds0 + ABUF * 2;
  const uint64_t gbase = (uint64_t)(uintptr_t)(xb + (size_t)tok * DIM + sc);

  v8f acc[4][4] = {};
  const __bf16* abase = As + (lane & 15) * APITCH + ((lane < 16) ? 0 : 16);
  const __bf16* bbase = w1b + ((size_t)(e * (DIM / 32)) * (HIDDEN / 16) + (n0 >> 4)) * 512
                            + lane * 16;
  const int NK = DIM / KCH;                      // 8
  async_copy64B(lds0, gbase);
  for (int kc = 0; kc < NK; ++kc) {
    if (kc) __syncthreads();                     // buf[(kc+1)&1] free for refill
    if (kc + 1 < NK) {
      async_copy64B((kc & 1) ? lds0 : lds1, gbase + (size_t)(kc + 1) * (KCH * 2));
      async_wait4();                             // chunk kc retired (in-order)
    } else {
      async_wait0();
    }
    __syncthreads();                             // all waves' staging visible
    const __bf16* ab  = abase + ((kc & 1) ? ABUF : 0);
    const __bf16* bkc = bbase + (size_t)(kc * 4) * ((HIDDEN / 16) * 512);
    for (int kb = 0; kb < 4; ++kb) {
      const __bf16* bp = bkc + (size_t)kb * ((HIDDEN / 16) * 512);
      __builtin_prefetch(bp + (HIDDEN / 16) * 512, 0, 3);
      bf16x16 bfr[4];
#pragma unroll
      for (int ni = 0; ni < 4; ++ni) bfr[ni] = ld_b_frag(bp + ni * 512);
#pragma unroll
      for (int mi = 0; mi < 4; ++mi) {
        const bf16x16 a = ld_a_frag(ab + mi * 16 * APITCH + kb * 32);
#pragma unroll
        for (int ni = 0; ni < 4; ++ni)
          acc[mi][ni] = __builtin_amdgcn_wmma_f32_16x16x32_bf16(
              false, a, false, bfr[ni], (short)0, acc[mi][ni], false, false);
      }
    }
  }
  // C layout: VGPR r -> M = r + (lane<16?0:8), N = lane&15
  const int mb  = (lane < 16) ? 0 : 8;
  const int nc0 = n0 + (lane & 15);
#pragma unroll
  for (int ni = 0; ni < 4; ++ni) {
    const int nc = nc0 + ni * 16;
    const float bias = b1[e * HIDDEN + nc];
#pragma unroll
    for (int mi = 0; mi < 4; ++mi) {
#pragma unroll
      for (int r = 0; r < 8; ++r) {
        const float v = acc[mi][ni][r] + bias;
        hbuf[(size_t)(tile * 64 + mi * 16 + mb + r) * HIDDEN + nc] =
            (__bf16)gelu_exact(v);
      }
    }
  }
}

// ---------------- GEMM2: out[token] = h @ w2[e] + b2[e]  (scatter fp32) -------
__global__ __launch_bounds__(256) void moe_gemm2(
    const __bf16* __restrict__ hbuf, const __bf16* __restrict__ w2b,
    const float* __restrict__ b2, const int* __restrict__ meta,
    const int* __restrict__ order, float* __restrict__ out) {
  __shared__ __bf16 As[2 * ABUF];
  const int tile = blockIdx.x;
  if (tile >= meta[12]) return;
  int e = 0;
#pragma unroll
  for (int q = 0; q < 3; ++q) if (tile >= meta[9 + q]) e = q + 1;
  const int ltile = tile - meta[8 + e];
  const int seg   = meta[4 + e];
  const int cnt   = meta[e];
  const int tid   = threadIdx.x;
  const int wave  = tid >> 5, lane = tid & 31;
  const int n0 = blockIdx.y * 512 + wave * 64;
  const int sr = tid >> 2;
  const int sc = (tid & 3) * 32;
  const uint32_t lds0 = (uint32_t)(uintptr_t)(As + sr * APITCH + sc);
  const uint32_t lds1 = lds0 + ABUF * 2;
  const uint64_t gbase =
      (uint64_t)(uintptr_t)(hbuf + (size_t)(tile * 64 + sr) * HIDDEN + sc);

  v8f acc[4][4] = {};
  const __bf16* abase = As + (lane & 15) * APITCH + ((lane < 16) ? 0 : 16);
  const __bf16* bbase = w2b + ((size_t)(e * (HIDDEN / 32)) * (DIM / 16) + (n0 >> 4)) * 512
                            + lane * 16;
  const int NK = HIDDEN / KCH;                   // 32
  async_copy64B(lds0, gbase);
  for (int kc = 0; kc < NK; ++kc) {
    if (kc) __syncthreads();
    if (kc + 1 < NK) {
      async_copy64B((kc & 1) ? lds0 : lds1, gbase + (size_t)(kc + 1) * (KCH * 2));
      async_wait4();
    } else {
      async_wait0();
    }
    __syncthreads();
    const __bf16* ab  = abase + ((kc & 1) ? ABUF : 0);
    const __bf16* bkc = bbase + (size_t)(kc * 4) * ((DIM / 16) * 512);
    for (int kb = 0; kb < 4; ++kb) {
      const __bf16* bp = bkc + (size_t)kb * ((DIM / 16) * 512);
      __builtin_prefetch(bp + (DIM / 16) * 512, 0, 3);
      bf16x16 bfr[4];
#pragma unroll
      for (int ni = 0; ni < 4; ++ni) bfr[ni] = ld_b_frag(bp + ni * 512);
#pragma unroll
      for (int mi = 0; mi < 4; ++mi) {
        const bf16x16 a = ld_a_frag(ab + mi * 16 * APITCH + kb * 32);
#pragma unroll
        for (int ni = 0; ni < 4; ++ni)
          acc[mi][ni] = __builtin_amdgcn_wmma_f32_16x16x32_bf16(
              false, a, false, bfr[ni], (short)0, acc[mi][ni], false, false);
      }
    }
  }
  const int mb  = (lane < 16) ? 0 : 8;
  const int nc0 = n0 + (lane & 15);
#pragma unroll
  for (int ni = 0; ni < 4; ++ni) {
    const int nc = nc0 + ni * 16;
    const float bias = b2[e * DIM + nc];
#pragma unroll
    for (int mi = 0; mi < 4; ++mi) {
#pragma unroll
      for (int r = 0; r < 8; ++r) {
        const int rl = ltile * 64 + mi * 16 + mb + r;
        if (rl < cnt) {
          const int tok = order[seg + rl];
          out[(size_t)tok * DIM + nc] = acc[mi][ni][r] + bias;
        }
      }
    }
  }
}

// ---------------- launch ----------------
extern "C" void kernel_launch(void* const* d_in, const int* in_sizes, int n_in,
                              void* d_out, int out_size, void* d_ws, size_t ws_size,
                              hipStream_t stream) {
  const float* x  = (const float*)d_in[0];
  const float* rw = (const float*)d_in[1];
  const float* rb = (const float*)d_in[2];
  const float* w1 = (const float*)d_in[3];
  const float* b1 = (const float*)d_in[4];
  const float* w2 = (const float*)d_in[5];
  const float* b2 = (const float*)d_in[6];
  float* out = (float*)d_out;
  char*  ws  = (char*)d_ws;   // requires ~146 MB workspace

  int*    meta  = (int*)(ws + WS_META);
  int*    idx   = (int*)(ws + WS_IDX);
  int*    order = (int*)(ws + WS_ORD);
  __bf16* xb    = (__bf16*)(ws + WS_XB);
  __bf16* w1b   = (__bf16*)(ws + WS_W1B);
  __bf16* w2b   = (__bf16*)(ws + WS_W2B);
  __bf16* hb    = (__bf16*)(ws + WS_H);

  moe_init<<<1, 32, 0, stream>>>(meta);
  moe_router<<<NTOK / 8, 256, 0, stream>>>(x, rw, rb, meta, idx);
  moe_scan<<<1, 1, 0, stream>>>(meta);
  moe_scatter<<<NTOK / 256, 256, 0, stream>>>(idx, meta, order);
  moe_prep_x<<<(NTOK * DIM) / 256, 256, 0, stream>>>(x, xb);
  moe_prep_w1<<<(NEXP * DIM * HIDDEN) / 256, 256, 0, stream>>>(w1, w1b);
  moe_prep_w2<<<(NEXP * HIDDEN * DIM) / 256, 256, 0, stream>>>(w2, w2b);
  dim3 g1(NTOK / 64 + NEXP, HIDDEN / 512);
  moe_gemm1<<<g1, 256, 0, stream>>>(xb, w1b, b1, meta, order, hb);
  dim3 g2(NTOK / 64 + NEXP, DIM / 512);
  moe_gemm2<<<g2, 256, 0, stream>>>(hb, w2b, b2, meta, order, out);
}
